// Model_30863634989256
// MI455X (gfx1250) — matrix operand order repaired and checked by reading, not compiled, not run on hardware
//
#include <hip/hip_runtime.h>

typedef _Float16 v16h __attribute__((ext_vector_type(16)));
typedef _Float16 v8h  __attribute__((ext_vector_type(8)));
typedef float    v8f  __attribute__((ext_vector_type(8)));

union V16 { v16h v; v8h h[2]; };

// ---- CDNA5 async LDS-copy path (toolchain-guarded; fallback = sync loads) ----
#if defined(__HIP_DEVICE_COMPILE__) && __has_builtin(__builtin_amdgcn_global_load_async_to_lds_b64)
#define HAS_ASYNC_LDS 1
typedef int v2i __attribute__((ext_vector_type(2)));
typedef __attribute__((address_space(1))) v2i* as1_v2i;
typedef __attribute__((address_space(3))) v2i* as3_v2i;
#define ASYNC_B64(g, l) \
  __builtin_amdgcn_global_load_async_to_lds_b64((as1_v2i)(void*)(g), (as3_v2i)(void*)(l), 0, 0)
#if __has_builtin(__builtin_amdgcn_s_wait_asynccnt)
#define WAIT_ASYNC(n) __builtin_amdgcn_s_wait_asynccnt(n)
#else
#define WAIT_ASYNC(n) asm volatile("s_wait_asynccnt %0" :: "i"(n) : "memory")
#endif
#else
#define HAS_ASYNC_LDS 0
#endif

// ---- problem constants ----
constexpr int BB  = 64;           // batch
constexpr int TT  = 2048;         // time
constexpr int FF  = 40;           // freq
constexpr int CF  = 8;            // conv filters
constexpr int Ho  = 2046;         // T - K + 1
constexpr int Wo  = 38;           // F - K + 1
constexpr int GIN = 304;          // Wo * CF
constexpr int GINP= 320;          // padded K for WMMA (10 x 32)
constexpr int GH  = 32;
constexpr int G3  = 96;           // 3 * GH
constexpr int ATT = 32;           // attention dim
constexpr long M  = (long)Ho * BB;        // 130944 GEMM rows
constexpr int  MT = (int)(M / 16);        // 8184 row tiles
constexpr int  NTg= G3 / 16;              // 6 col tiles
constexpr int  KBg= GINP / 32;            // 10 K blocks

// ---- d_out offsets (floats, reference tuple order) ----
constexpr long O_LOGITS = 0;
constexpr long O_SUM    = O_LOGITS + (long)BB*2;
constexpr long O_MUL    = O_SUM    + (long)BB*GH;
constexpr long O_SM     = O_MUL    + (long)BB*Ho*GH;
constexpr long O_L2     = O_SM     + (long)BB*Ho*GH;
constexpr long O_GRU    = O_L2     + (long)BB*Ho*GH;
constexpr long O_DCN    = O_GRU    + (long)Ho*BB*GH;
constexpr long O_DCNW   = O_DCN    + (long)BB*CF*Ho*Wo;
constexpr long O_DCNB   = O_DCNW   + 72;

// ---- workspace offsets (bytes) ----
constexpr size_t WS_HA16 = 0;                                   // M * 320 f16
constexpr size_t WS_GI   = WS_HA16 + (size_t)M*GINP*2;          // M * 96 f32
constexpr size_t WS_ENC  = WS_GI   + (size_t)M*G3*4;            // M * 32 f16
constexpr size_t WS_WIHP = WS_ENC  + (size_t)M*GH*2;            // 6*10*32*16 f16
constexpr size_t WS_WHHP = WS_WIHP + 30720*2;                   // 6*32*16 f16
constexpr size_t WS_W1P  = WS_WHHP + 3072*2;                    // 2*32*16 f16
constexpr size_t WS_W2P  = WS_W1P  + 1024*2;                    // 2*32*16 f16

__device__ __forceinline__ float sigm(float x){ return 1.0f/(1.0f+__expf(-x)); }

// =====================================================================
// 1) Pack weights into WMMA B-fragment order (+ copy dcn_w/dcn_b outputs)
// =====================================================================
__global__ void k_pack(const float* __restrict__ w_ih, const float* __restrict__ w_hh,
                       const float* __restrict__ w1,   const float* __restrict__ w2,
                       const float* __restrict__ dcnw, const float* __restrict__ dcnb,
                       _Float16* __restrict__ wihP, _Float16* __restrict__ whhP,
                       _Float16* __restrict__ w1P,  _Float16* __restrict__ w2P,
                       float* __restrict__ o_dcnw, float* __restrict__ o_dcnb)
{
  int i = blockIdx.x * blockDim.x + threadIdx.x;
  if (i < 30720) {                                    // w_ih: 6 ntiles x 10 kb
    int e = i & 15, lane = (i >> 4) & 31, kb = (i >> 9) % KBg, nt = i / (512*KBg);
    int k = kb*32 + ((lane < 16) ? 0 : 8) + (e & 7) + ((e & 8) ? 16 : 0);
    int n = nt*16 + (lane & 15);
    wihP[i] = (_Float16)((k < GIN) ? w_ih[n*GIN + k] : 0.0f);
  } else if (i < 30720 + 3072) {                      // w_hh: 6 ntiles, K=32
    int j = i - 30720;
    int e = j & 15, lane = (j >> 4) & 31, nt = j >> 9;
    int k = ((lane < 16) ? 0 : 8) + (e & 7) + ((e & 8) ? 16 : 0);
    int n = nt*16 + (lane & 15);
    whhP[j] = (_Float16)w_hh[n*GH + k];
  } else if (i < 30720 + 3072 + 1024) {               // w1: 2 ntiles
    int j = i - 30720 - 3072;
    int e = j & 15, lane = (j >> 4) & 31, nt = j >> 9;
    int k = ((lane < 16) ? 0 : 8) + (e & 7) + ((e & 8) ? 16 : 0);
    int n = nt*16 + (lane & 15);
    w1P[j] = (_Float16)w1[n*GH + k];
  } else if (i < 30720 + 3072 + 2048) {               // w2: 2 ntiles
    int j = i - 30720 - 3072 - 1024;
    int e = j & 15, lane = (j >> 4) & 31, nt = j >> 9;
    int k = ((lane < 16) ? 0 : 8) + (e & 7) + ((e & 8) ? 16 : 0);
    int n = nt*16 + (lane & 15);
    w2P[j] = (_Float16)w2[n*ATT + k];
  } else if (i < 35840 + 72) {
    o_dcnw[i - 35840] = dcnw[i - 35840];
  } else if (i < 35912 + 8) {
    o_dcnb[i - 35912] = dcnb[i - 35912];
  }
}

// =====================================================================
// 2) Fused offset-conv + bilinear deformable sampling + tap mix.
// =====================================================================
__global__ void k_deform(const float* __restrict__ x,
                         const float* __restrict__ offw, const float* __restrict__ offb,
                         const float* __restrict__ dcnw, const float* __restrict__ dcnb,
                         float* __restrict__ out_dcn, _Float16* __restrict__ hA16)
{
  __shared__ float s_offw[162];
  __shared__ float s_offb[18];
  __shared__ float s_dcnw[72];
  __shared__ float s_dcnb[8];
  int tid = threadIdx.x;
  if (tid < 162) s_offw[tid] = offw[tid];
  if (tid < 18)  s_offb[tid] = offb[tid];
  if (tid < 72)  s_dcnw[tid] = dcnw[tid];
  if (tid < 8)   s_dcnb[tid] = dcnb[tid];
  __syncthreads();

  long gid = (long)blockIdx.x * blockDim.x + tid;
  const long total = (long)BB * Ho * Wo;
  if (gid >= total) return;
  int w = (int)(gid % Wo);
  int t = (int)((gid / Wo) % Ho);
  int b = (int)(gid / ((long)Wo * Ho));
  const float* img = x + (long)b * TT * FF;

  float off[18];
  #pragma unroll
  for (int c = 0; c < 18; ++c) off[c] = s_offb[c];
  #pragma unroll
  for (int i = 0; i < 3; ++i)
    #pragma unroll
    for (int j = 0; j < 3; ++j) {
      float xv = img[(t + i) * FF + (w + j)];
      #pragma unroll
      for (int c = 0; c < 18; ++c) off[c] += xv * s_offw[c * 9 + i * 3 + j];
    }

  float samp[9];
  #pragma unroll
  for (int tap = 0; tap < 9; ++tap) {
    float ph = off[tap]     + (float)t + (float)(tap / 3);
    float pw = off[9 + tap] + (float)w + (float)(tap % 3);
    ph = fminf(fmaxf(ph, 0.0f), (float)(TT - 1));
    pw = fminf(fmaxf(pw, 0.0f), (float)(FF - 1));
    int h0 = (int)floorf(ph), w0 = (int)floorf(pw);
    int h1 = min(h0 + 1, TT - 1), w1 = min(w0 + 1, FF - 1);
    float lh = ph - (float)h0, lw = pw - (float)w0;
    float v00 = img[h0 * FF + w0], v01 = img[h0 * FF + w1];
    float v10 = img[h1 * FF + w0], v11 = img[h1 * FF + w1];
    samp[tap] = v00*(1.f-lh)*(1.f-lw) + v01*(1.f-lh)*lw + v10*lh*(1.f-lw) + v11*lh*lw;
  }

  long rowOff = ((long)t * BB + b) * GINP + (long)w * CF;
  #pragma unroll
  for (int cf = 0; cf < CF; ++cf) {
    float o = s_dcnb[cf];
    #pragma unroll
    for (int tap = 0; tap < 9; ++tap) o += samp[tap] * s_dcnw[cf * 9 + tap];
    out_dcn[(((long)b * CF + cf) * Ho + t) * Wo + w] = o;
    hA16[rowOff + cf] = (_Float16)o;
  }
  if (w == 0) {
    long pr = ((long)t * BB + b) * GINP;
    #pragma unroll
    for (int k = GIN; k < GINP; ++k) hA16[pr + k] = (_Float16)0.0f;
  }
}

// =====================================================================
// 3) Big GEMM: gi[M x 96] = hA16[M x 320] * w_ih^T + b_ih  (WMMA f16->f32)
// =====================================================================
__global__ void k_gemm_gi(const _Float16* __restrict__ A,
                          const _Float16* __restrict__ Bp,
                          const float* __restrict__ b_ih,
                          float* __restrict__ gi)
{
  int wid  = (int)((blockIdx.x * blockDim.x + threadIdx.x) >> 5);
  int lane = threadIdx.x & 31;
  int mt = wid / NTg, nt = wid % NTg;
  if (mt >= MT) return;                       // wave-uniform guard
  long row  = (long)mt * 16 + (lane & 15);
  int  sel  = (lane < 16) ? 0 : 8;
  const _Float16* arow = A + row * GINP;
  v8f c = {0.f,0.f,0.f,0.f,0.f,0.f,0.f,0.f};
  #pragma unroll 2
  for (int kb = 0; kb < KBg; ++kb) {
    if (kb + 1 < KBg) __builtin_prefetch(arow + (kb + 1) * 32, 0, 3);
    V16 a, bm;
    const _Float16* ap = arow + kb * 32 + sel;
    a.h[0] = *(const v8h*)(ap);
    a.h[1] = *(const v8h*)(ap + 16);
    bm.v   = *(const v16h*)(Bp + ((long)(nt * KBg + kb) * 32 + lane) * 16);
    c = __builtin_amdgcn_wmma_f32_16x16x32_f16(false, a.v, false, bm.v,
                                               (short)0, c, false, false);
  }
  int col = nt * 16 + (lane & 15);
  float bias = b_ih[col];
  long rbase = (long)mt * 16 + ((lane < 16) ? 0 : 8);
  #pragma unroll
  for (int r = 0; r < 8; ++r)
    gi[(rbase + r) * G3 + col] = c[r] + bias;
}

// =====================================================================
// 4) GRU scan. 4 blocks; block handles 16 batch rows. Per step:
//    gh(16x96) = h16(16x32) x w_hh^T via 6 WMMAs (waves 0..5), then gates.
//    gi[t] tiles are double-buffered into LDS with CDNA5 async-LDS loads
//    (ASYNCcnt) when the toolchain exposes them; else read synchronously.
// =====================================================================
__global__ void k_gru(const float* __restrict__ gi,
                      const _Float16* __restrict__ whhP,
                      const float* __restrict__ b_hh,
                      float* __restrict__ out_gru,
                      _Float16* __restrict__ enc16)
{
  __shared__ __align__(32) float    sh[16 * GH];
  __shared__ __align__(32) _Float16 sh16[16 * GH];
  __shared__ __align__(32) float    sgh[16 * G3];
  __shared__ __align__(32) _Float16 swhh[NTg * 32 * 16];
  __shared__ float sbhh[G3];
#if HAS_ASYNC_LDS
  __shared__ __align__(16) float sgi[2][16 * G3];     // double-buffered gi tile
#endif

  int tid = threadIdx.x;
  int bbase = blockIdx.x * 16;
  for (int i = tid; i < NTg * 32 * 16; i += blockDim.x) swhh[i] = whhP[i];
  if (tid < G3) sbhh[tid] = b_hh[tid];
  for (int i = tid; i < 16 * GH; i += blockDim.x) { sh[i] = 0.0f; sh16[i] = (_Float16)0.0f; }

#if HAS_ASYNC_LDS
  // prologue: async-stage gi tile for t=0 (16 rows x 96 f32, contiguous 6KB)
  {
    const float* src = gi + ((long)0 * BB + bbase) * G3;
    #pragma unroll
    for (int kch = 0; kch < 3; ++kch) {
      int i = tid + kch * 256;                        // 768 B64 chunks
      ASYNC_B64(src + i * 2, &sgi[0][i * 2]);
    }
    WAIT_ASYNC(0);
  }
#endif
  __syncthreads();

  int lane = tid & 31;
  int wv   = tid >> 5;
  int sel  = (lane < 16) ? 0 : 8;

  for (int t = 0; t < Ho; ++t) {
#if HAS_ASYNC_LDS
    if (t + 1 < Ho) {                                 // stage next tile async
      const float* src = gi + ((long)(t + 1) * BB + bbase) * G3;
      float* dst = &sgi[(t + 1) & 1][0];
      #pragma unroll
      for (int kch = 0; kch < 3; ++kch) {
        int i = tid + kch * 256;
        ASYNC_B64(src + i * 2, dst + i * 2);
      }
    }
#endif
    if (wv < NTg) {                                   // wave-uniform; EXEC full
      V16 a, bm;
      const _Float16* ap = sh16 + (lane & 15) * GH + sel;
      a.h[0] = *(const v8h*)(ap);
      a.h[1] = *(const v8h*)(ap + 16);
      bm.v   = *(const v16h*)(swhh + (wv * 32 + lane) * 16);
      v8f c = {0.f,0.f,0.f,0.f,0.f,0.f,0.f,0.f};
      c = __builtin_amdgcn_wmma_f32_16x16x32_f16(false, a.v, false, bm.v,
                                                 (short)0, c, false, false);
      int col = wv * 16 + (lane & 15);
      int rb  = (lane < 16) ? 0 : 8;
      #pragma unroll
      for (int r = 0; r < 8; ++r) sgh[(rb + r) * G3 + col] = c[r];
    }
#if HAS_ASYNC_LDS
    if (t + 1 < Ho) WAIT_ASYNC(3);                    // oldest 3 (tile t) done
    else            WAIT_ASYNC(0);
#endif
    __syncthreads();
    for (int e = tid; e < 16 * GH; e += blockDim.x) {
      int brow = e >> 5, g = e & 31;
      int b = bbase + brow;
#if HAS_ASYNC_LDS
      const float* git = &sgi[t & 1][brow * G3];
      float ir  = git[g];
      float iz  = git[GH + g];
      float inn = git[2 * GH + g];
#else
      long gbase = ((long)t * BB + b) * G3;
      float ir  = gi[gbase + g];
      float iz  = gi[gbase + GH + g];
      float inn = gi[gbase + 2 * GH + g];
#endif
      float hr  = sgh[brow * G3 + g]          + sbhh[g];
      float hz  = sgh[brow * G3 + GH + g]     + sbhh[GH + g];
      float hn  = sgh[brow * G3 + 2 * GH + g] + sbhh[2 * GH + g];
      float r = sigm(ir + hr);
      float z = sigm(iz + hz);
      float n = tanhf(inn + r * hn);
      float hnew = (1.0f - z) * n + z * sh[e];
      sh[e]   = hnew;
      sh16[e] = (_Float16)hnew;
      long oi = ((long)t * BB + b) * GH + g;
      out_gru[oi] = hnew;
      enc16[oi]   = (_Float16)hnew;
    }
    __syncthreads();
  }
}

// =====================================================================
// 5) Fused attention MLP: l2 = tanh(enc @ w1^T + b1) @ w2^T + b2.
// =====================================================================
__global__ void k_attn(const _Float16* __restrict__ enc16,
                       const _Float16* __restrict__ w1P, const float* __restrict__ b1,
                       const _Float16* __restrict__ w2P, const float* __restrict__ b2,
                       float* __restrict__ out_l2)
{
  __shared__ __align__(32) _Float16 sa[8][16 * GH];   // per-wave scratch tile
  int tid = threadIdx.x, lane = tid & 31, wv = tid >> 5;
  int wid = blockIdx.x * 8 + wv;
  int b  = wid >> 7;            // / 128 t-tiles
  int tt = wid & 127;
  int sel  = (lane < 16) ? 0 : 8;
  int colb = lane & 15;
  int rb   = (lane < 16) ? 0 : 8;
  int trow = tt * 16 + colb;

  V16 a;
  if (trow < Ho) {
    const _Float16* ap = enc16 + ((long)trow * BB + b) * GH + sel;
    a.h[0] = *(const v8h*)(ap);
    a.h[1] = *(const v8h*)(ap + 16);
  } else {
    for (int i = 0; i < 16; ++i) a.v[i] = (_Float16)0.0f;
  }

  _Float16* my = &sa[wv][0];
  #pragma unroll
  for (int nt = 0; nt < 2; ++nt) {
    V16 bm; bm.v = *(const v16h*)(w1P + (nt * 32 + lane) * 16);
    v8f c = {0.f,0.f,0.f,0.f,0.f,0.f,0.f,0.f};
    c = __builtin_amdgcn_wmma_f32_16x16x32_f16(false, a.v, false, bm.v,
                                               (short)0, c, false, false);
    int col = nt * 16 + colb;
    #pragma unroll
    for (int r = 0; r < 8; ++r)
      my[(rb + r) * GH + col] = (_Float16)tanhf(c[r] + b1[col]);
  }
  // restage: LDS ops are in-order within a wave
  V16 a2;
  const _Float16* ap2 = my + colb * GH + sel;
  a2.h[0] = *(const v8h*)(ap2);
  a2.h[1] = *(const v8h*)(ap2 + 16);
  #pragma unroll
  for (int nt = 0; nt < 2; ++nt) {
    V16 bm; bm.v = *(const v16h*)(w2P + (nt * 32 + lane) * 16);
    v8f c = {0.f,0.f,0.f,0.f,0.f,0.f,0.f,0.f};
    c = __builtin_amdgcn_wmma_f32_16x16x32_f16(false, a2.v, false, bm.v,
                                               (short)0, c, false, false);
    int col = nt * 16 + colb;
    float bias = b2[col];
    #pragma unroll
    for (int r = 0; r < 8; ++r) {
      int tg = tt * 16 + rb + r;
      if (tg < Ho)
        out_l2[((long)b * Ho + tg) * GH + col] = c[r] + bias;
    }
  }
}

// =====================================================================
// 6) Softmax over time + weighted sum + logits. One block per batch row.
// =====================================================================
__global__ void k_softmax(const float* __restrict__ l2, const float* __restrict__ gru,
                          const float* __restrict__ w3, const float* __restrict__ b3,
                          float* __restrict__ out_sm, float* __restrict__ out_mul,
                          float* __restrict__ out_sum, float* __restrict__ logits)
{
  int b = blockIdx.x, tid = threadIdx.x;
  int g = tid & 31, slice = tid >> 5;           // 8 time slices
  __shared__ float red[8][GH];
  __shared__ float gmax[GH], gsum[GH], osum[GH];
  const float* l2b = l2 + (long)b * Ho * GH;

  float m = -3.4e38f;
  for (int t = slice; t < Ho; t += 8) m = fmaxf(m, l2b[t * GH + g]);
  red[slice][g] = m;
  __syncthreads();
  if (slice == 0) {
    float mm = red[0][g];
    #pragma unroll
    for (int s = 1; s < 8; ++s) mm = fmaxf(mm, red[s][g]);
    gmax[g] = mm;
  }
  __syncthreads();
  float mx = gmax[g];

  float s = 0.0f;
  for (int t = slice; t < Ho; t += 8) s += __expf(l2b[t * GH + g] - mx);
  red[slice][g] = s;
  __syncthreads();
  if (slice == 0) {
    float ss = 0.0f;
    #pragma unroll
    for (int q = 0; q < 8; ++q) ss += red[q][g];
    gsum[g] = ss;
  }
  __syncthreads();
  float inv = 1.0f / gsum[g];

  float acc = 0.0f;
  for (int t = slice; t < Ho; t += 8) {
    float smv = __expf(l2b[t * GH + g] - mx) * inv;
    float ev  = gru[((long)t * BB + b) * GH + g];
    long oi = (long)b * Ho * GH + (long)t * GH + g;
    out_sm[oi]  = smv;
    float mult  = smv * ev;
    out_mul[oi] = mult;
    acc += mult;
  }
  red[slice][g] = acc;
  __syncthreads();
  if (slice == 0) {
    float ss = 0.0f;
    #pragma unroll
    for (int q = 0; q < 8; ++q) ss += red[q][g];
    osum[g] = ss;
    out_sum[b * GH + g] = ss;
  }
  __syncthreads();
  if (tid < 2) {
    float lg = b3[tid];
    #pragma unroll
    for (int k = 0; k < GH; ++k) lg += osum[k] * w3[tid * GH + k];
    logits[b * 2 + tid] = lg;
  }
}

// =====================================================================
extern "C" void kernel_launch(void* const* d_in, const int* in_sizes, int n_in,
                              void* d_out, int out_size, void* d_ws, size_t ws_size,
                              hipStream_t stream) {
  const float* x     = (const float*)d_in[0];
  const float* offw  = (const float*)d_in[1];
  const float* offb  = (const float*)d_in[2];
  const float* dcnw  = (const float*)d_in[3];
  const float* dcnb  = (const float*)d_in[4];
  const float* w_ih  = (const float*)d_in[5];
  const float* w_hh  = (const float*)d_in[6];
  const float* b_ih  = (const float*)d_in[7];
  const float* b_hh  = (const float*)d_in[8];
  const float* w1    = (const float*)d_in[9];
  const float* b1    = (const float*)d_in[10];
  const float* w2    = (const float*)d_in[11];
  const float* b2    = (const float*)d_in[12];
  const float* w3    = (const float*)d_in[13];
  const float* b3    = (const float*)d_in[14];

  float* out = (float*)d_out;
  char*  ws  = (char*)d_ws;
  _Float16* hA16  = (_Float16*)(ws + WS_HA16);
  float*    gi    = (float*)   (ws + WS_GI);
  _Float16* enc16 = (_Float16*)(ws + WS_ENC);
  _Float16* wihP  = (_Float16*)(ws + WS_WIHP);
  _Float16* whhP  = (_Float16*)(ws + WS_WHHP);
  _Float16* w1P   = (_Float16*)(ws + WS_W1P);
  _Float16* w2P   = (_Float16*)(ws + WS_W2P);

  // 1) pack weights + copy dcn_w/dcn_b outputs
  k_pack<<<141, 256, 0, stream>>>(w_ih, w_hh, w1, w2, dcnw, dcnb,
                                  wihP, whhP, w1P, w2P,
                                  out + O_DCNW, out + O_DCNB);
  // 2) deformable conv (writes out_dcn + hA16)
  k_deform<<<19437, 256, 0, stream>>>(x, offw, offb, dcnw, dcnb,
                                      out + O_DCN, hA16);
  // 3) big input-gate GEMM (WMMA)
  k_gemm_gi<<<(MT * NTg) / 8, 256, 0, stream>>>(hA16, wihP, b_ih, gi);
  // 4) sequential GRU scan (WMMA per step + async-LDS double buffering)
  k_gru<<<4, 256, 0, stream>>>(gi, whhP, b_hh, out + O_GRU, enc16);
  // 5) fused attention MLP (WMMA x4 per tile)
  k_attn<<<(BB * 128) / 8, 256, 0, stream>>>(enc16, w1P, b1, w2P, b2, out + O_L2);
  // 6) softmax over time + weighted sum + logits
  k_softmax<<<BB, 256, 0, stream>>>(out + O_L2, out + O_GRU, w3, b3,
                                    out + O_SM, out + O_MUL,
                                    out + O_SUM, out + O_LOGITS);
}
